// TriFoldBlock_4629974745262
// MI455X (gfx1250) — matrix operand order repaired and checked by reading, not compiled, not run on hardware
//
#include <hip/hip_runtime.h>
#include <cstdint>
#include <cstddef>

// ---------------------------------------------------------------------------
// TriFoldBlock for MI455X (gfx1250, wave32, WMMA 16x16x32 f16->f32).
//   x: (1,256,256,128) f32.  All GEMMs via v_wmma_f32_16x16x32_f16.
// ---------------------------------------------------------------------------

#define LSZ  256
#define DSZ  128
#define NTOT (LSZ * LSZ) /* 65536 rows */

typedef _Float16 half_t;
typedef __attribute__((ext_vector_type(16))) _Float16 v16h;
typedef __attribute__((ext_vector_type(8)))  _Float16 v8h;
typedef __attribute__((ext_vector_type(4)))  _Float16 v4h;
typedef __attribute__((ext_vector_type(8)))  float    v8f;
typedef __attribute__((ext_vector_type(4)))  float    f4;

union V16U { v16h v; v8h h[2]; };

__device__ __forceinline__ v8f wmma_f16(v16h a, v16h b, v8f c) {
  return __builtin_amdgcn_wmma_f32_16x16x32_f16(false, a, false, b, (short)0, c,
                                                false, false);
}

// A fragment (16x32 f16): lane = hi*16 + m; element e -> A[m][hi*16+e]
// Source must have K contiguous per row; two 16B loads per lane.
__device__ __forceinline__ v16h load_a_frag(const half_t* base, int ld, int lane) {
  int m = lane & 15, hi = (lane >> 4) & 1;
  const half_t* p = base + m * ld + hi * 16;
  V16U u;
  u.h[0] = *(const v8h*)p;
  u.h[1] = *(const v8h*)(p + 8);
  return u.v;
}

// B fragment (32x16 f16): lane = hi*16 + n; element e -> B[hi*16+e][n]
// Source must have K contiguous per column (column stride = ldc halves).
__device__ __forceinline__ v16h load_b_frag(const half_t* base, int ldc, int lane) {
  int n = lane & 15, hi = (lane >> 4) & 1;
  const half_t* p = base + n * ldc + hi * 16;
  V16U u;
  u.h[0] = *(const v8h*)p;
  u.h[1] = *(const v8h*)(p + 8);
  return u.v;
}

// Pre-packed weight fragment: 512 halves per (kchunk, ntile); lane-contiguous.
__device__ __forceinline__ v16h load_w_frag(const half_t* wp, int lane) {
  const half_t* p = wp + lane * 16;
  V16U u;
  u.h[0] = *(const v8h*)p;
  u.h[1] = *(const v8h*)(p + 8);
  return u.v;
}

__device__ __forceinline__ float sigmoid_f(float x) {
  return 1.0f / (1.0f + __expf(-x));
}

// ---------------------------------------------------------------------------
// Weight packing: f32 W[K x N] row-major -> f16 B-fragments.
// frag index = kchunk*ntiles + ntile;  out[frag*512 + lane*16 + e].
// ---------------------------------------------------------------------------
__global__ __launch_bounds__(32)
void pack_w_kernel(const float* __restrict__ W, half_t* __restrict__ out,
                   int ncols, int ntiles) {
  int blk = blockIdx.x, lane = threadIdx.x;
  int kt = blk / ntiles, nt = blk % ntiles;
  int n = nt * 16 + (lane & 15);
  int hi = (lane >> 4) & 1;
  half_t* o = out + (size_t)blk * 512 + lane * 16;
#pragma unroll
  for (int e = 0; e < 16; ++e) {
    int k = kt * 32 + hi * 16 + e;
    o[e] = (half_t)W[(size_t)k * ncols + n];
  }
}

// ---------------------------------------------------------------------------
// LayerNorm over D=128 per row; one wave per row; f16 output row-major.
// transposed: write at row' = (j*256+i) for input row (i*256+j).
// ---------------------------------------------------------------------------
__global__ __launch_bounds__(256)
void ln_kernel(const float* __restrict__ x, const float* __restrict__ g,
               const float* __restrict__ bb, half_t* __restrict__ out,
               int transposed) {
  int lane = threadIdx.x & 31;
  int row = blockIdx.x * 8 + (threadIdx.x >> 5);
  const float* xr = x + (size_t)row * DSZ;
  f4 v = *(const f4*)(xr + lane * 4);
  float s  = v[0] + v[1] + v[2] + v[3];
  float s2 = v[0]*v[0] + v[1]*v[1] + v[2]*v[2] + v[3]*v[3];
#pragma unroll
  for (int m = 1; m < 32; m <<= 1) {
    s  += __shfl_xor(s, m);
    s2 += __shfl_xor(s2, m);
  }
  float mean = s * (1.0f / DSZ);
  float var  = fmaxf(s2 * (1.0f / DSZ) - mean * mean, 0.0f);
  float inv  = rsqrtf(var + 1e-5f);
  int orow = transposed ? (((row & 255) << 8) | (row >> 8)) : row;
  v4h o;
#pragma unroll
  for (int c = 0; c < 4; ++c) {
    int col = lane * 4 + c;
    o[c] = (half_t)((v[c] - mean) * inv * g[col] + bb[col]);
  }
  *(v4h*)(out + (size_t)orow * DSZ + lane * 4) = o;
}

// f32 -> f16 row-major copy (for gate-GEMM input).
__global__ __launch_bounds__(256)
void f2h_kernel(const float* __restrict__ in, half_t* __restrict__ out) {
  size_t i = ((size_t)blockIdx.x * 256 + threadIdx.x) * 4;
  f4 v = *(const f4*)(in + i);
  v4h o;
  o[0] = (half_t)v[0]; o[1] = (half_t)v[1];
  o[2] = (half_t)v[2]; o[3] = (half_t)v[3];
  *(v4h*)(out + i) = o;
}

// ---------------------------------------------------------------------------
// Gated projection for trimul: P = (z@W + b) * sigmoid(z@Wg + bg)
// Output packed channel-major: out[h][row]; packT swaps (i,k)->(k,i) in row.
// ---------------------------------------------------------------------------
__global__ __launch_bounds__(32)
void proj_gated_pack_kernel(const half_t* __restrict__ zh,
                            const half_t* __restrict__ wp,  const float* __restrict__ b1,
                            const half_t* __restrict__ wpg, const float* __restrict__ bg,
                            half_t* __restrict__ outp, int packT) {
  int lane = threadIdx.x;
  int rt = blockIdx.x, ct = blockIdx.y;
  v8f acc = {}, accg = {};
#pragma unroll
  for (int kc = 0; kc < 4; ++kc) {
    v16h a  = load_a_frag(zh + (size_t)rt * 16 * DSZ + kc * 32, DSZ, lane);
    v16h w  = load_w_frag(wp  + (size_t)(kc * 8 + ct) * 512, lane);
    v16h wg = load_w_frag(wpg + (size_t)(kc * 8 + ct) * 512, lane);
    acc  = wmma_f16(a, w,  acc);
    accg = wmma_f16(a, wg, accg);
  }
  int n = lane & 15, hi = (lane >> 4) & 1;
  int col = ct * 16 + n;
  int row0 = rt * 16 + hi * 8;
  float bv = b1[col], bgv = bg[col];
  if (!packT) {
    v8h o;
#pragma unroll
    for (int r = 0; r < 8; ++r)
      o[r] = (half_t)((acc[r] + bv) * sigmoid_f(accg[r] + bgv));
    *(v8h*)(outp + (size_t)col * NTOT + row0) = o;
  } else {
#pragma unroll
    for (int r = 0; r < 8; ++r) {
      int row = row0 + r;
      int rp = ((row & 255) << 8) | (row >> 8);
      outp[(size_t)col * NTOT + rp] =
          (half_t)((acc[r] + bv) * sigmoid_f(accg[r] + bgv));
    }
  }
}

// ---------------------------------------------------------------------------
// Plain projection (q/k/v, transition-W1): out = act(z@W + b), K = 128.
// PACKED: channel-major out[col][row]; else row-major (ld = ncols).
// ---------------------------------------------------------------------------
template <int PACKED, int RELU>
__global__ __launch_bounds__(32)
void proj_kernel(const half_t* __restrict__ zh, const half_t* __restrict__ wp,
                 const float* __restrict__ bias, half_t* __restrict__ out,
                 int ncols, int ntiles) {
  int lane = threadIdx.x;
  int rt = blockIdx.x, ct = blockIdx.y;
  v8f acc = {};
#pragma unroll
  for (int kc = 0; kc < 4; ++kc) {
    v16h a = load_a_frag(zh + (size_t)rt * 16 * DSZ + kc * 32, DSZ, lane);
    v16h w = load_w_frag(wp + (size_t)(kc * ntiles + ct) * 512, lane);
    acc = wmma_f16(a, w, acc);
  }
  int n = lane & 15, hi = (lane >> 4) & 1;
  int col = ct * 16 + n;
  int row0 = rt * 16 + hi * 8;
  float bv = bias ? bias[col] : 0.0f;
  if (PACKED) {
    v8h o;
#pragma unroll
    for (int r = 0; r < 8; ++r) {
      float vv = acc[r] + bv;
      if (RELU) vv = fmaxf(vv, 0.0f);
      o[r] = (half_t)vv;
    }
    *(v8h*)(out + (size_t)col * NTOT + row0) = o;
  } else {
#pragma unroll
    for (int r = 0; r < 8; ++r) {
      float vv = acc[r] + bv;
      if (RELU) vv = fmaxf(vv, 0.0f);
      out[(size_t)(row0 + r) * ncols + col] = (half_t)vv;
    }
  }
}

// ---------------------------------------------------------------------------
// Trimul core: core[i,j,h] = sum_k Lp[h][i*256+k] * Rp[h][j*256+k]   (K=256)
// (incoming variant handled by transposed packing upstream)
// ---------------------------------------------------------------------------
__global__ __launch_bounds__(32)
void trimul_core_kernel(const half_t* __restrict__ lp,
                        const half_t* __restrict__ rp,
                        float* __restrict__ core) {
  int lane = threadIdx.x;
  int it = blockIdx.x, jt = blockIdx.y, h = blockIdx.z;
  const half_t* lb = lp + (size_t)h * NTOT;
  const half_t* rb = rp + (size_t)h * NTOT;
  v8f acc = {};
#pragma unroll
  for (int kc = 0; kc < 8; ++kc) {
    v16h a = load_a_frag(lb + (size_t)it * 16 * LSZ + kc * 32, LSZ, lane);
    v16h b = load_b_frag(rb + (size_t)jt * 16 * LSZ + kc * 32, LSZ, lane);
    acc = wmma_f16(a, b, acc);
  }
  int n = lane & 15, hi = (lane >> 4) & 1;
  float* cp = core + ((size_t)(it * 16 + hi * 8) * LSZ + jt * 16 + n) * DSZ + h;
#pragma unroll
  for (int r = 0; r < 8; ++r) cp[(size_t)r * LSZ * DSZ] = acc[r];
}

// ---------------------------------------------------------------------------
// Flash attention per (q-tile 16, row x, head). head_dim = 32 == WMMA K.
// qh/kh: row-major f16 (ld=128); vp: channel-major f16 vp[c][x*256+kpos].
// Output oh row-major f16; transposed un-swaps rows for the "ending" node.
// ---------------------------------------------------------------------------
__global__ __launch_bounds__(32)
void flash_attn_kernel(const half_t* __restrict__ qh, const half_t* __restrict__ kh,
                       const half_t* __restrict__ vp, half_t* __restrict__ oh,
                       int transposed) {
  __shared__ __align__(16) half_t pl[16 * 32];
  int lane = threadIdx.x;
  int qt = blockIdx.x, xr = blockIdx.y, h = blockIdx.z;
  int n = lane & 15, hi = (lane >> 4) & 1;
  const float scale = 0.1767766952966369f; // 1/sqrt(32)

  v16h qa = load_a_frag(qh + ((size_t)(xr * LSZ + qt * 16)) * DSZ + h * 32, DSZ, lane);
  v8f o0 = {}, o1 = {};
  float mrow[8], lrow[8];
#pragma unroll
  for (int r = 0; r < 8; ++r) { mrow[r] = -1e30f; lrow[r] = 0.0f; }

  for (int kc = 0; kc < 8; ++kc) {
    int kb = kc * 32;
    v16h k0 = load_b_frag(kh + ((size_t)(xr * LSZ + kb)) * DSZ + h * 32, DSZ, lane);
    v16h k1 = load_b_frag(kh + ((size_t)(xr * LSZ + kb + 16)) * DSZ + h * 32, DSZ, lane);
    v8f s0 = {}, s1 = {};
    s0 = wmma_f16(qa, k0, s0);
    s1 = wmma_f16(qa, k1, s1);
#pragma unroll
    for (int r = 0; r < 8; ++r) {
      float a0 = s0[r] * scale, a1 = s1[r] * scale;
      float mx = fmaxf(a0, a1);
      mx = fmaxf(mx, __shfl_xor(mx, 1));
      mx = fmaxf(mx, __shfl_xor(mx, 2));
      mx = fmaxf(mx, __shfl_xor(mx, 4));
      mx = fmaxf(mx, __shfl_xor(mx, 8));
      float mnew = fmaxf(mrow[r], mx);
      float alpha = __expf(mrow[r] - mnew);
      mrow[r] = mnew;
      float p0 = __expf(a0 - mnew), p1 = __expf(a1 - mnew);
      float rs = p0 + p1;
      rs += __shfl_xor(rs, 1);
      rs += __shfl_xor(rs, 2);
      rs += __shfl_xor(rs, 4);
      rs += __shfl_xor(rs, 8);
      lrow[r] = lrow[r] * alpha + rs;
      o0[r] *= alpha;
      o1[r] *= alpha;
      pl[(hi * 8 + r) * 32 + n]      = (half_t)p0;
      pl[(hi * 8 + r) * 32 + n + 16] = (half_t)p1;
    }
    __syncthreads();
    v16h pa = load_a_frag(pl, 32, lane);
    v16h v0 = load_b_frag(vp + (size_t)(h * 32) * NTOT + xr * LSZ + kb, NTOT, lane);
    v16h v1 = load_b_frag(vp + (size_t)(h * 32 + 16) * NTOT + xr * LSZ + kb, NTOT, lane);
    o0 = wmma_f16(pa, v0, o0);
    o1 = wmma_f16(pa, v1, o1);
    __syncthreads();
  }
#pragma unroll
  for (int r = 0; r < 8; ++r) {
    float inv = 1.0f / lrow[r];
    int q = qt * 16 + hi * 8 + r;
    int row = transposed ? (q * LSZ + xr) : (xr * LSZ + q);
    half_t* op = oh + (size_t)row * DSZ + h * 32 + n;
    op[0]  = (half_t)(o0[r] * inv);
    op[16] = (half_t)(o1[r] * inv);
  }
}

// ---------------------------------------------------------------------------
// Residual epilogue:
//   GATED:  x += sigmoid(xh@Wg + bg) * (ah@Wo + bo)      (KDIM = 128)
//   plain:  x += ah@W2 + b2                              (KDIM = 512)
// In-place f32 update of x.
// ---------------------------------------------------------------------------
template <int GATED, int KDIM>
__global__ __launch_bounds__(32)
void resid_kernel(const half_t* __restrict__ ah, const half_t* __restrict__ wp,
                  const float* __restrict__ b1,
                  const half_t* __restrict__ xh, const half_t* __restrict__ wpg,
                  const float* __restrict__ bg,
                  float* __restrict__ xf) {
  int lane = threadIdx.x;
  int rt = blockIdx.x, ct = blockIdx.y;
  v8f acc = {}, accg = {};
#pragma unroll
  for (int kc = 0; kc < KDIM / 32; ++kc) {
    v16h a = load_a_frag(ah + (size_t)rt * 16 * KDIM + kc * 32, KDIM, lane);
    v16h w = load_w_frag(wp + (size_t)(kc * 8 + ct) * 512, lane);
    acc = wmma_f16(a, w, acc);
  }
  if (GATED) {
#pragma unroll
    for (int kc = 0; kc < 4; ++kc) {
      v16h a = load_a_frag(xh + (size_t)rt * 16 * DSZ + kc * 32, DSZ, lane);
      v16h w = load_w_frag(wpg + (size_t)(kc * 8 + ct) * 512, lane);
      accg = wmma_f16(a, w, accg);
    }
  }
  int n = lane & 15, hi = (lane >> 4) & 1;
  int col = ct * 16 + n;
  float bv = b1[col];
  float bgv = GATED ? bg[col] : 0.0f;
#pragma unroll
  for (int r = 0; r < 8; ++r) {
    size_t idx = (size_t)(rt * 16 + hi * 8 + r) * DSZ + col;
    float zo = acc[r] + bv;
    float val = GATED ? (xf[idx] + sigmoid_f(accg[r] + bgv) * zo)
                      : (xf[idx] + zo);
    xf[idx] = val;
  }
}

// ---------------------------------------------------------------------------
// Host orchestration.
// Input order (setup_inputs dict order, flattened):
//   0: x
//   1..16:  mul_out  {ng nb Wl bl Wr br Wlg blg Wrg brg ong onb Wo bo Wg bg}
//   17..32: mul_in   (same order)
//   33..41: att_start {ng nb Wq Wk Wv Wo bo Wg bg}
//   42..50: att_end   (same order)
//   51..56: transition {ng nb W1 b1 W2 b2}
// Workspace (~113 MB): zh(16M) xh(16M) A(16M) B(16M) C(16M) core(32M) wpack(1M)
// transition hidden (64M) aliases A..C(+part of core), which are free then.
// ---------------------------------------------------------------------------
extern "C" void kernel_launch(void* const* d_in, const int* in_sizes, int n_in,
                              void* d_out, int out_size, void* d_ws, size_t ws_size,
                              hipStream_t stream) {
  (void)in_sizes; (void)n_in; (void)out_size; (void)ws_size;
  auto F = [&](int i) { return (const float*)d_in[i]; };
  float* xf = (float*)d_out;

  char* w = (char*)d_ws;
  const size_t MB = 1u << 20;
  half_t* zh   = (half_t*)(w + 0 * MB);
  half_t* xh   = (half_t*)(w + 16 * MB);
  half_t* bufA = (half_t*)(w + 32 * MB);
  half_t* bufB = (half_t*)(w + 48 * MB);
  half_t* bufC = (half_t*)(w + 64 * MB);
  float*  core = (float*) (w + 80 * MB);
  half_t* hh   = bufA; // 64 MB transition hidden aliases bufA..bufC(+core head)
  half_t* WP   = (half_t*)(w + 112 * MB);
  auto slot = [&](int s) { return WP + (size_t)s * 16384; };

  (void)hipMemcpyAsync(xf, d_in[0], (size_t)NTOT * DSZ * sizeof(float),
                       hipMemcpyDeviceToDevice, stream);

  auto pack = [&](const float* Wm, half_t* dst, int K, int Nc) {
    int kt = K / 32, nt = Nc / 16;
    pack_w_kernel<<<dim3(kt * nt), dim3(32), 0, stream>>>(Wm, dst, Nc, nt);
  };

  int muls[2] = {1, 17};
  for (int m = 0; m < 2; ++m) {
    int b = muls[m], s = m * 6;
    pack(F(b + 2),  slot(s + 0), 128, 128); // Wl
    pack(F(b + 6),  slot(s + 1), 128, 128); // Wlg
    pack(F(b + 4),  slot(s + 2), 128, 128); // Wr
    pack(F(b + 8),  slot(s + 3), 128, 128); // Wrg
    pack(F(b + 12), slot(s + 4), 128, 128); // Wo
    pack(F(b + 14), slot(s + 5), 128, 128); // Wg
  }
  int atts[2] = {33, 42};
  for (int m = 0; m < 2; ++m) {
    int b = atts[m], s = 12 + m * 5;
    pack(F(b + 2), slot(s + 0), 128, 128); // Wq
    pack(F(b + 3), slot(s + 1), 128, 128); // Wk
    pack(F(b + 4), slot(s + 2), 128, 128); // Wv
    pack(F(b + 5), slot(s + 3), 128, 128); // Wo
    pack(F(b + 7), slot(s + 4), 128, 128); // Wg
  }
  pack(F(53), slot(22), 128, 512); // W1 (4 slots)
  pack(F(55), slot(26), 512, 128); // W2 (4 slots)

  dim3 gGemm(NTOT / 16, 8), blk32(32), blk256(256);
  dim3 gLN(NTOT / 8);
  dim3 gCore(16, 16, 128);
  dim3 gFlash(16, LSZ, 4);

  // ---- triangle multiplications (outgoing, then incoming) ----
  for (int m = 0; m < 2; ++m) {
    int b = muls[m], s = m * 6;
    int packT = m; // incoming: reduction over first L axis -> transposed pack
    ln_kernel<<<gLN, blk256, 0, stream>>>(xf, F(b + 0), F(b + 1), zh, 0);
    proj_gated_pack_kernel<<<gGemm, blk32, 0, stream>>>(
        zh, slot(s + 0), F(b + 3), slot(s + 1), F(b + 7), bufA, packT);
    proj_gated_pack_kernel<<<gGemm, blk32, 0, stream>>>(
        zh, slot(s + 2), F(b + 5), slot(s + 3), F(b + 9), bufB, packT);
    trimul_core_kernel<<<gCore, blk32, 0, stream>>>(bufA, bufB, core);
    ln_kernel<<<gLN, blk256, 0, stream>>>(core, F(b + 10), F(b + 11), zh, 0);
    f2h_kernel<<<gLN, blk256, 0, stream>>>(xf, xh);
    resid_kernel<1, 128><<<gGemm, blk32, 0, stream>>>(
        zh, slot(s + 4), F(b + 13), xh, slot(s + 5), F(b + 15), xf);
  }

  // ---- triangle attentions (starting, then ending) ----
  for (int m = 0; m < 2; ++m) {
    int b = atts[m], s = 12 + m * 5;
    int tr = m;
    ln_kernel<<<gLN, blk256, 0, stream>>>(xf, F(b + 0), F(b + 1), zh, tr);
    proj_kernel<0, 0><<<gGemm, blk32, 0, stream>>>(zh, slot(s + 0), nullptr, bufA, 128, 8); // q
    proj_kernel<0, 0><<<gGemm, blk32, 0, stream>>>(zh, slot(s + 1), nullptr, bufB, 128, 8); // k
    proj_kernel<1, 0><<<gGemm, blk32, 0, stream>>>(zh, slot(s + 2), nullptr, bufC, 128, 8); // v packed
    flash_attn_kernel<<<gFlash, blk32, 0, stream>>>(bufA, bufB, bufC, zh, tr);  // oh -> zh
    f2h_kernel<<<gLN, blk256, 0, stream>>>(xf, xh);
    resid_kernel<1, 128><<<gGemm, blk32, 0, stream>>>(
        zh, slot(s + 3), F(b + 6), xh, slot(s + 4), F(b + 8), xf);
  }

  // ---- transition ----
  {
    int b = 51;
    ln_kernel<<<gLN, blk256, 0, stream>>>(xf, F(b + 0), F(b + 1), zh, 0);
    proj_kernel<0, 1><<<dim3(NTOT / 16, 32), blk32, 0, stream>>>(
        zh, slot(22), F(b + 3), hh, 512, 32); // relu(z@W1+b1), 65536x512
    resid_kernel<0, 512><<<gGemm, blk32, 0, stream>>>(
        hh, slot(26), F(b + 5), nullptr, nullptr, nullptr, xf);
  }
}
